// LinearSecondOrderMimo_9208409882674
// MI455X (gfx1250) — matrix operand order repaired and checked by reading, not compiled, not run on hardware
//
#include <hip/hip_runtime.h>
#include <math.h>

typedef __attribute__((ext_vector_type(16))) _Float16 v16h;
typedef __attribute__((ext_vector_type(8)))  _Float16 v8h;
typedef __attribute__((ext_vector_type(8)))  float    v8f;
typedef __attribute__((ext_vector_type(4)))  unsigned int u32x4;
typedef __attribute__((ext_vector_type(4)))  int      i32x4;
typedef __attribute__((ext_vector_type(8)))  int      i32x8;

#define KTAPS   128                 // truncated impulse-response length (r^128 << f32 eps)
#define TCHUNK  256                 // timesteps per workgroup
#define UROWS   (TCHUNK + KTAPS)    // staged input window rows (384)
#define NTHR    256                 // 8 wave32
#define OI      256                 // O*I = 16*16

#if defined(__AMDGCN__) && __has_builtin(__builtin_amdgcn_tensor_load_to_lds)
#define USE_TDM 1
#else
#define USE_TDM 0
#endif

// ---------------------------------------------------------------------------
// Prologue: per-(o,i) 2nd-order impulse response, stored as hi/lo f16 split
// in layout H[k][o][i]  (k-major so WMMA A-fragments are contiguous 16B rows).
// Hh and Hl are contiguous: one 128KB block -> single TDM descriptor.
// ---------------------------------------------------------------------------
__global__ void soi_impulse_kernel(const float* __restrict__ b_coeff,  // (O,I,2)
                                   const float* __restrict__ rho,      // (O,I,1)
                                   const float* __restrict__ psi,      // (O,I,1)
                                   _Float16* __restrict__ Hh,
                                   _Float16* __restrict__ Hl)
{
    int idx = threadIdx.x;               // idx = o*16 + i
    if (idx >= OI) return;

    float r     = 1.0f / (1.0f + expf(-rho[idx]));
    float theta = 3.14159265358979323846f / (1.0f + expf(-psi[idx]));  // pi*sigmoid
    float a1    = -2.0f * r * cosf(theta);
    float a2    = r * r;
    float b0    = b_coeff[2 * idx + 0];
    float b1    = b_coeff[2 * idx + 1];

    float y1 = 0.0f, y2 = 0.0f;
    for (int k = 0; k < KTAPS; ++k) {
        float x = (k == 0) ? b0 : ((k == 1) ? b1 : 0.0f);
        float h = x - a1 * y1 - a2 * y2;   // impulse response of full biquad
        y2 = y1; y1 = h;
        _Float16 hh = (_Float16)h;
        _Float16 hl = (_Float16)(h - (float)hh);   // error-compensation term
        Hh[k * OI + idx] = hh;
        Hl[k * OI + idx] = hl;
    }
}

// ---------------------------------------------------------------------------
// Main GEMM: out[(b,t),o] = sum_{k,i} H[o,(k,i)] * u[b,t-k,i]  via WMMA f16
// with hi/lo compensated products accumulating in f32.
// ---------------------------------------------------------------------------
__global__ __launch_bounds__(NTHR)
void soi_wmma_kernel(const float* __restrict__ u,        // (B,T,16) f32
                     const _Float16* __restrict__ Hh_g,  // Hh..Hl contiguous, 128KB
                     float* __restrict__ out, int T)     // (B,T,16) f32
{
    extern __shared__ _Float16 smem[];
    _Float16* sHh = smem;                        // KTAPS*OI halves (64 KB)
    _Float16* sHl = sHh + KTAPS * OI;            // 64 KB (contiguous with sHh)
    _Float16* sUh = sHl + KTAPS * OI;            // UROWS*16 halves (12 KB)
    _Float16* sUl = sUh + UROWS * 16;            // 12 KB

    const int b   = blockIdx.y;
    const int tb  = blockIdx.x * TCHUNK;
    const int tid = threadIdx.x;

    // ---- stage impulse-response matrices into LDS ----
#if USE_TDM
    if (tid < 32) {
        // Tensor DMA: one 1-D descriptor moving 2*KTAPS*OI halves = 128 KB,
        // as 16384 x 8-byte elements.  D# layout per CDNA5 ISA ch.8.
        const unsigned long long ga = (unsigned long long)(uintptr_t)Hh_g;
        const unsigned lds_off = 0;                   // sHh at base of LDS alloc
        const unsigned nelem = (2 * KTAPS * OI * 2) / 8;   // 16384

        u32x4 g0 = {};
        g0.x = 1u;                                    // count=1 valid descriptor
        g0.y = lds_off;                               // lds_addr [63:32]
        g0.z = (unsigned)(ga & 0xFFFFFFFFull);        // global_addr [95:64]
        g0.w = (unsigned)((ga >> 32) & 0x01FFFFFFull) // global_addr [120:96]
             | (2u << 30);                            // type=2 ("image") [127:126]

        i32x8 g1 = {};
        g1[0] = (3 << 16);                            // data_size=3 (8B), mask=0
        g1[1] = (int)((nelem & 0xFFFFu) << 16);       // tensor_dim0 [79:48] lo16
        g1[2] = (int)(((nelem >> 16) & 0xFFFFu)       // tensor_dim0 hi16
             | (1u << 16));                           // tensor_dim1 = 1
        g1[3] = (int)((nelem & 0xFFFFu) << 16);       // tile_dim0 [127:112]
        g1[4] = 1;                                    // tile_dim1 = 1, tile_dim2 = 0
        g1[5] = (int)nelem;                           // tensor_dim0_stride lo32
        g1[6] = 0;                                    // stride0 hi16, stride1 lo16
        g1[7] = 0;

        i32x4 gz = {};
#if __clang_major__ >= 23
        i32x8 gz8 = {};
        __builtin_amdgcn_tensor_load_to_lds(g0, g1, gz, gz, gz8, 0);
#else
        __builtin_amdgcn_tensor_load_to_lds(g0, g1, gz, gz, 0);
#endif
        __builtin_amdgcn_s_wait_tensorcnt(0);
    }
#else
    {
        const v8h* src = (const v8h*)Hh_g;
        v8h* dst = (v8h*)sHh;
        for (int v = tid; v < (2 * KTAPS * OI) / 8; v += NTHR)
            dst[v] = src[v];
    }
#endif

    // ---- stage shifted input window, f32 -> (hi,lo) f16 split (VALU work,
    //      overlaps with the TDM transfer above) ----
    {
        const float* ub = u + (size_t)b * T * 16;
        for (int e = tid; e < UROWS * 16; e += NTHR) {
            int j = e >> 4;                     // window row
            int i = e & 15;                     // input channel
            int t = tb + j - KTAPS;             // global timestep (zero-pad t<0)
            float val = (t >= 0) ? ub[t * 16 + i] : 0.0f;
            _Float16 vh = (_Float16)val;
            sUh[e] = vh;
            sUl[e] = (_Float16)(val - (float)vh);
        }
    }
    __syncthreads();

    const int lane = tid & 31;
    const int wave = tid >> 5;        // 0..7
    const int n    = lane & 15;       // A: row m (=o) ; B/D: column (=t offset)
    const int hi   = lane >> 4;       // half-wave select
    const int i0   = hi * 8;
    const int tloc = wave * 32;       // each wave owns two 16-step tiles

    v8f acc0 = {};
    v8f acc1 = {};

    #pragma unroll 2
    for (int c = 0; c < KTAPS / 2; ++c) {
        const int k0 = 2 * c;

        // A fragment (16x32 f16, contiguous K = e + hi*16):
        //   elems 0..7  = H[k0+hi][m=n][0..7]
        //   elems 8..15 = H[k0+hi][m=n][8..15]
        v8h a0h = *(const v8h*)&sHh[(k0 + hi) * OI + n * 16 + 0];
        v8h a1h = *(const v8h*)&sHh[(k0 + hi) * OI + n * 16 + 8];
        v8h a0l = *(const v8h*)&sHl[(k0 + hi) * OI + n * 16 + 0];
        v8h a1l = *(const v8h*)&sHl[(k0 + hi) * OI + n * 16 + 8];
        v16h ah = __builtin_shufflevector(a0h, a1h, 0,1,2,3,4,5,6,7,8,9,10,11,12,13,14,15);
        v16h al = __builtin_shufflevector(a0l, a1l, 0,1,2,3,4,5,6,7,8,9,10,11,12,13,14,15);

        // B fragment (32x16 f16): lane col n, elem e -> K = e + hi*16
        //   -> tap (k0+hi), channel e  => one contiguous 16-channel u row.
        const int j0 = tloc + n + KTAPS - k0 - hi;   // tile-0 window row
        const int j1 = j0 + 16;                       // tile-1 window row
        v8h u0ha = *(const v8h*)&sUh[j0 * 16 + 0];
        v8h u0hb = *(const v8h*)&sUh[j0 * 16 + 8];
        v8h u0la = *(const v8h*)&sUl[j0 * 16 + 0];
        v8h u0lb = *(const v8h*)&sUl[j0 * 16 + 8];
        v8h u1ha = *(const v8h*)&sUh[j1 * 16 + 0];
        v8h u1hb = *(const v8h*)&sUh[j1 * 16 + 8];
        v8h u1la = *(const v8h*)&sUl[j1 * 16 + 0];
        v8h u1lb = *(const v8h*)&sUl[j1 * 16 + 8];
        v16h bh0 = __builtin_shufflevector(u0ha, u0hb, 0,1,2,3,4,5,6,7,8,9,10,11,12,13,14,15);
        v16h bl0 = __builtin_shufflevector(u0la, u0lb, 0,1,2,3,4,5,6,7,8,9,10,11,12,13,14,15);
        v16h bh1 = __builtin_shufflevector(u1ha, u1hb, 0,1,2,3,4,5,6,7,8,9,10,11,12,13,14,15);
        v16h bl1 = __builtin_shufflevector(u1la, u1lb, 0,1,2,3,4,5,6,7,8,9,10,11,12,13,14,15);

        // hi*hi + hi*lo + lo*hi  (lo*lo ~ 2^-22, dropped); f32 accumulate.
        acc0 = __builtin_amdgcn_wmma_f32_16x16x32_f16(false, ah, false, bh0, (short)0, acc0, false, false);
        acc0 = __builtin_amdgcn_wmma_f32_16x16x32_f16(false, ah, false, bl0, (short)0, acc0, false, false);
        acc0 = __builtin_amdgcn_wmma_f32_16x16x32_f16(false, al, false, bh0, (short)0, acc0, false, false);
        acc1 = __builtin_amdgcn_wmma_f32_16x16x32_f16(false, ah, false, bh1, (short)0, acc1, false, false);
        acc1 = __builtin_amdgcn_wmma_f32_16x16x32_f16(false, ah, false, bl1, (short)0, acc1, false, false);
        acc1 = __builtin_amdgcn_wmma_f32_16x16x32_f16(false, al, false, bh1, (short)0, acc1, false, false);
    }

    // D layout: lane holds column t = base + n, rows o = i0..i0+7 in v8f -> one
    // contiguous 32B store per tile.
    {
        const int t0g = tb + tloc + n;
        float* p0 = out + ((size_t)b * T + t0g) * 16 + i0;
        *(v8f*)p0 = acc0;
        const int t1g = t0g + 16;
        float* p1 = out + ((size_t)b * T + t1g) * 16 + i0;
        *(v8f*)p1 = acc1;
    }
}

// ---------------------------------------------------------------------------
extern "C" void kernel_launch(void* const* d_in, const int* in_sizes, int n_in,
                              void* d_out, int out_size, void* d_ws, size_t ws_size,
                              hipStream_t stream) {
    (void)n_in; (void)out_size; (void)ws_size;

    const float* u_in    = (const float*)d_in[0];   // (B,T,16)
    const float* b_coeff = (const float*)d_in[1];   // (16,16,2)
    const float* rho     = (const float*)d_in[2];   // (16,16,1)
    const float* psi     = (const float*)d_in[3];   // (16,16,1)
    float*       out     = (float*)d_out;           // (B,T,16)

    const int B = 32;
    const int T = in_sizes[0] / (B * 16);           // 4096 in the reference

    _Float16* Hh = (_Float16*)d_ws;                 // 64 KB
    _Float16* Hl = Hh + KTAPS * OI;                 // 64 KB, contiguous after Hh

    soi_impulse_kernel<<<dim3(1), dim3(256), 0, stream>>>(b_coeff, rho, psi, Hh, Hl);

    const size_t smem_bytes = (size_t)(2 * KTAPS * OI + 2 * UROWS * 16) * sizeof(_Float16); // 152 KB
    (void)hipFuncSetAttribute((const void*)soi_wmma_kernel,
                              hipFuncAttributeMaxDynamicSharedMemorySize,
                              (int)smem_bytes);

    dim3 grid(T / TCHUNK, B);
    soi_wmma_kernel<<<grid, dim3(NTHR), smem_bytes, stream>>>(u_in, Hh, out, T);
}
